// MemoryAttention_31851477467510
// MI455X (gfx1250) — matrix operand order repaired and checked by reading, not compiled, not run on hardware
//
#include <hip/hip_runtime.h>

// ---------------------------------------------------------------------------
// MemoryAttention for MI455X (gfx1250, wave32, WMMA bf16 path)
// B=2, Q=1024, K=2048, M=1024, E=1024, H=32, HD=32
// ---------------------------------------------------------------------------

typedef __bf16 bf16;
typedef __attribute__((ext_vector_type(16))) __bf16 v16bf;
typedef __attribute__((ext_vector_type(8)))  __bf16 v8bf;
typedef __attribute__((ext_vector_type(8)))  float  v8f;

#define Bd   2
#define Qd   1024
#define Kd   2048
#define Md   1024
#define Ed   1024
#define Hd   32
#define HDd  32
#define SCALEF 0.17677669529663687f   // 1/sqrt(32)

// ---- WMMA fragment loaders (contiguous-k order, same for A and B, wave32) ----------
// A (16x32 bf16, MxK): lane L holds row M=L%16; K = (L/16)*16 .. +15 contiguous
__device__ __forceinline__ v16bf load_a_frag(const bf16* base, int ld) {
  const int lane = threadIdx.x & 31;
  const bf16* p = base + (size_t)(lane & 15) * ld + ((lane >> 4) << 4);
  v8bf lo = *(const v8bf*)(p);
  v8bf hi = *(const v8bf*)(p + 8);
  v16bf a;
#pragma unroll
  for (int i = 0; i < 8; ++i) { a[i] = lo[i]; a[i + 8] = hi[i]; }
  return a;
}

// B (32x16 bf16, KxN): lane L holds col N=L%16; K = (L/16)*16 .. +15 contiguous.
// Source is row-major [N][K-contiguous] with leading dim ld -> one 32B vector load.
__device__ __forceinline__ v16bf load_b_frag(const bf16* base, int ld) {
  const int lane = threadIdx.x & 31;
  return *(const v16bf*)(base + (size_t)(lane & 15) * ld + ((lane >> 4) << 4));
}

// ---- 16-lane butterflies via DS_SWIZZLE_B32 (group-of-32 xor mode) ---------
// offset encoding: [15]=0, xor_mask=[14:10], or_mask=[9:5], and_mask=[4:0]
template <int PAT>
__device__ __forceinline__ float ds_swz(float v) {
  return __int_as_float(__builtin_amdgcn_ds_swizzle(__float_as_int(v), PAT));
}
__device__ __forceinline__ float rowmax16(float v) {
  v = fmaxf(v, ds_swz<0x041F>(v));   // xor 1
  v = fmaxf(v, ds_swz<0x081F>(v));   // xor 2
  v = fmaxf(v, ds_swz<0x101F>(v));   // xor 4
  v = fmaxf(v, ds_swz<0x201F>(v));   // xor 8
  return v;
}
__device__ __forceinline__ float rowsum16(float v) {
  v += ds_swz<0x041F>(v);
  v += ds_swz<0x081F>(v);
  v += ds_swz<0x101F>(v);
  v += ds_swz<0x201F>(v);
  return v;
}

// ---- f32 -> bf16 conversion ------------------------------------------------
__global__ void __launch_bounds__(256) cvt_f32_bf16(const float* __restrict__ in,
                                                    bf16* __restrict__ out, int n) {
  int i = blockIdx.x * 256 + threadIdx.x;
  if (i < n) out[i] = (bf16)in[i];
}

// ---- GEMM: Y[r,f] = sum_e A[r,e]*W[f,e] + bias[f] --------------------------
// block = 128 threads (4 waves); block tile 64x64; wave tile 32x32 (2x2 WMMA).
// MODE 0: f32 row-major (B*L, E) output (final projection)
// MODE 1: bf16 head-major  out[((b*H+h)*L + l)*HD + d]
// MODE 2: bf16 head-transposed out[((b*H+h)*HD + d)*L + l]   (for V)
template <int MODE>
__global__ void __launch_bounds__(128)
gemm_bf16_k(const bf16* __restrict__ A, const bf16* __restrict__ W,
            const float* __restrict__ bias, void* __restrict__ out, int Lseq) {
  const int lane = threadIdx.x & 31;
  const int wid  = threadIdx.x >> 5;
  const int row0 = blockIdx.y * 64 + (wid >> 1) * 32;
  const int col0 = blockIdx.x * 64 + (wid & 1) * 32;

  v8f acc[2][2] = {};
  for (int kb = 0; kb < Ed; kb += 32) {
    v16bf a0 = load_a_frag(A + (size_t)(row0 +  0) * Ed + kb, Ed);
    v16bf a1 = load_a_frag(A + (size_t)(row0 + 16) * Ed + kb, Ed);
    v16bf w0 = load_b_frag(W + (size_t)(col0 +  0) * Ed + kb, Ed);
    v16bf w1 = load_b_frag(W + (size_t)(col0 + 16) * Ed + kb, Ed);
    acc[0][0] = __builtin_amdgcn_wmma_f32_16x16x32_bf16(false, a0, false, w0, (short)0, acc[0][0], false, false);
    acc[0][1] = __builtin_amdgcn_wmma_f32_16x16x32_bf16(false, a0, false, w1, (short)0, acc[0][1], false, false);
    acc[1][0] = __builtin_amdgcn_wmma_f32_16x16x32_bf16(false, a1, false, w0, (short)0, acc[1][0], false, false);
    acc[1][1] = __builtin_amdgcn_wmma_f32_16x16x32_bf16(false, a1, false, w1, (short)0, acc[1][1], false, false);
  }

  // D layout: vgpr i, lanes 0-15 -> row i, lanes 16-31 -> row i+8; col = lane%16
  const int half = lane >> 4, nl = lane & 15;
#pragma unroll
  for (int mt = 0; mt < 2; ++mt)
#pragma unroll
    for (int nt = 0; nt < 2; ++nt) {
      const int f = col0 + nt * 16 + nl;
      const float bv = bias[f];
#pragma unroll
      for (int i = 0; i < 8; ++i) {
        const int r = row0 + mt * 16 + i + half * 8;
        const float v = acc[mt][nt][i] + bv;
        if (MODE == 0) {
          ((float*)out)[(size_t)r * Ed + f] = v;
        } else {
          const int bb = r / Lseq, l = r - bb * Lseq;
          const int h = f >> 5, d = f & 31;
          const size_t idx = (MODE == 1)
              ? ((((size_t)bb * Hd + h) * Lseq + l) * HDd + d)
              : ((((size_t)bb * Hd + h) * HDd + d) * (size_t)Lseq + l);
          ((bf16*)out)[idx] = (bf16)v;
        }
      }
    }
}

// ---- Flash attention over concatenated [K ; M] keys ------------------------
// 1 wave per 16-row Q tile; 64-key chunks (4 score WMMAs + 4 PV WMMAs each).
// qp:(B,H,Q,HD) kp:(B,H,K,HD) vT:(B,H,HD,K) mkp:(B,H,M,HD) mvT:(B,H,HD,M)
// out:(B,Q,E) bf16
__global__ void __launch_bounds__(32)
attn_flash_k(const bf16* __restrict__ qp, const bf16* __restrict__ kp,
             const bf16* __restrict__ vT, const bf16* __restrict__ mkp,
             const bf16* __restrict__ mvT, bf16* __restrict__ outp) {
  const int lane = threadIdx.x & 31;
  const int qt = blockIdx.x, h = blockIdx.y, b = blockIdx.z;
  const int bh = b * Hd + h;

  __shared__ __align__(64) bf16 Pst[16 * 64];   // P staging: D-layout -> A-layout

  const v16bf qf = load_a_frag(qp + ((size_t)bh * Qd + qt * 16) * HDd, HDd);

  v8f o0 = {}, o1 = {};
  float rmax[8], rsum[8];
#pragma unroll
  for (int i = 0; i < 8; ++i) { rmax[i] = -__builtin_inff(); rsum[i] = 0.f; }

  const int half = lane >> 4, nl = lane & 15;
  const v8f z = {};

  for (int cb = 0; cb < Kd + Md; cb += 64) {
    const bf16 *kbase, *vbase;
    int seq, off;
    if (cb < Kd) { kbase = kp  + (size_t)bh * Kd * HDd; vbase = vT  + (size_t)bh * HDd * Kd; seq = Kd; off = cb; }
    else         { kbase = mkp + (size_t)bh * Md * HDd; vbase = mvT + (size_t)bh * HDd * Md; seq = Md; off = cb - Kd; }

    // scores: S(16x64) = Q(16x32) x K_chunk^T, as four 16x16 WMMA tiles
    v8f s0 = __builtin_amdgcn_wmma_f32_16x16x32_bf16(false, qf, false,
              load_b_frag(kbase + (size_t)(off +  0) * HDd, HDd), (short)0, z, false, false);
    v8f s1 = __builtin_amdgcn_wmma_f32_16x16x32_bf16(false, qf, false,
              load_b_frag(kbase + (size_t)(off + 16) * HDd, HDd), (short)0, z, false, false);
    v8f s2 = __builtin_amdgcn_wmma_f32_16x16x32_bf16(false, qf, false,
              load_b_frag(kbase + (size_t)(off + 32) * HDd, HDd), (short)0, z, false, false);
    v8f s3 = __builtin_amdgcn_wmma_f32_16x16x32_bf16(false, qf, false,
              load_b_frag(kbase + (size_t)(off + 48) * HDd, HDd), (short)0, z, false, false);

    // online softmax per row (row = vgpr i + 8*half; cols striped across 16 lanes)
#pragma unroll
    for (int i = 0; i < 8; ++i) {
      const float x0 = s0[i] * SCALEF, x1 = s1[i] * SCALEF;
      const float x2 = s2[i] * SCALEF, x3 = s3[i] * SCALEF;
      float m = rowmax16(fmaxf(fmaxf(x0, x1), fmaxf(x2, x3)));
      m = fmaxf(m, rmax[i]);
      const float alpha = __expf(rmax[i] - m);
      rmax[i] = m;
      const float p0 = __expf(x0 - m), p1 = __expf(x1 - m);
      const float p2 = __expf(x2 - m), p3 = __expf(x3 - m);
      rsum[i] = rsum[i] * alpha + rowsum16((p0 + p1) + (p2 + p3));
      o0[i] *= alpha; o1[i] *= alpha;
      const int r = i + half * 8;
      Pst[r * 64 + nl]      = (bf16)p0;
      Pst[r * 64 + nl + 16] = (bf16)p1;
      Pst[r * 64 + nl + 32] = (bf16)p2;
      Pst[r * 64 + nl + 48] = (bf16)p3;
    }
    __syncthreads();

    // O(16x32) += P(16x64) x V_chunk(64x32): A from LDS, B = contiguous vT rows
    v16bf pf0 = load_a_frag(Pst,      64);  // key cols  0..31
    v16bf pf1 = load_a_frag(Pst + 32, 64);  // key cols 32..63
    o0 = __builtin_amdgcn_wmma_f32_16x16x32_bf16(false, pf0, false,
           load_b_frag(vbase + off,                       seq), (short)0, o0, false, false);
    o0 = __builtin_amdgcn_wmma_f32_16x16x32_bf16(false, pf1, false,
           load_b_frag(vbase + off + 32,                  seq), (short)0, o0, false, false);
    o1 = __builtin_amdgcn_wmma_f32_16x16x32_bf16(false, pf0, false,
           load_b_frag(vbase + (size_t)16 * seq + off,      seq), (short)0, o1, false, false);
    o1 = __builtin_amdgcn_wmma_f32_16x16x32_bf16(false, pf1, false,
           load_b_frag(vbase + (size_t)16 * seq + off + 32, seq), (short)0, o1, false, false);
    __syncthreads();
  }

  // normalize and store to (B,Q,E) bf16
#pragma unroll
  for (int i = 0; i < 8; ++i) {
    const int r = qt * 16 + i + half * 8;
    const float inv = 1.0f / rsum[i];
    const size_t base = ((size_t)b * Qd + r) * Ed + h * HDd;
    outp[base + nl]      = (bf16)(o0[i] * inv);
    outp[base + 16 + nl] = (bf16)(o1[i] * inv);
  }
}

// ---------------------------------------------------------------------------
extern "C" void kernel_launch(void* const* d_in, const int* in_sizes, int n_in,
                              void* d_out, int out_size, void* d_ws, size_t ws_size,
                              hipStream_t stream) {
  const float* query  = (const float*)d_in[0];
  const float* key    = (const float*)d_in[1];
  const float* value  = (const float*)d_in[2];
  const float* memory = (const float*)d_in[3];
  const float* Wq = (const float*)d_in[4];  const float* bq = (const float*)d_in[5];
  const float* Wk = (const float*)d_in[6];  const float* bk = (const float*)d_in[7];
  const float* Wv = (const float*)d_in[8];  const float* bv = (const float*)d_in[9];
  const float* Wo = (const float*)d_in[10]; const float* bo = (const float*)d_in[11];

  // workspace carve-up (bf16), total = 32M elements = 64 MB
  bf16* p = (bf16*)d_ws;
  bf16* qbf  = p; p += (size_t)Bd * Qd * Ed;
  bf16* kbf  = p; p += (size_t)Bd * Kd * Ed;
  bf16* vbf  = p; p += (size_t)Bd * Kd * Ed;
  bf16* mbf  = p; p += (size_t)Bd * Md * Ed;
  bf16* wqb  = p; p += (size_t)Ed * Ed;
  bf16* wkb  = p; p += (size_t)Ed * Ed;
  bf16* wvb  = p; p += (size_t)Ed * Ed;
  bf16* wob  = p; p += (size_t)Ed * Ed;
  bf16* qproj  = p; p += (size_t)Bd * Hd * Qd * HDd;
  bf16* kproj  = p; p += (size_t)Bd * Hd * Kd * HDd;
  bf16* vTp    = p; p += (size_t)Bd * Hd * HDd * Kd;
  bf16* mkproj = p; p += (size_t)Bd * Hd * Md * HDd;
  bf16* mvTp   = p; p += (size_t)Bd * Hd * HDd * Md;
  bf16* attnb  = p; p += (size_t)Bd * Qd * Ed;

  auto cvt = [&](const float* in, bf16* out, size_t n) {
    cvt_f32_bf16<<<dim3((unsigned)((n + 255) / 256)), dim3(256), 0, stream>>>(in, out, (int)n);
  };
  cvt(query,  qbf, (size_t)Bd * Qd * Ed);
  cvt(key,    kbf, (size_t)Bd * Kd * Ed);
  cvt(value,  vbf, (size_t)Bd * Kd * Ed);
  cvt(memory, mbf, (size_t)Bd * Md * Ed);
  cvt(Wq, wqb, (size_t)Ed * Ed);
  cvt(Wk, wkb, (size_t)Ed * Ed);
  cvt(Wv, wvb, (size_t)Ed * Ed);
  cvt(Wo, wob, (size_t)Ed * Ed);

  const dim3 blk(128);
  // projections into head layouts (V/MV transposed for contiguous B-fragments)
  gemm_bf16_k<1><<<dim3(Ed / 64, (Bd * Qd) / 64), blk, 0, stream>>>(qbf, wqb, bq, qproj,  Qd);
  gemm_bf16_k<1><<<dim3(Ed / 64, (Bd * Kd) / 64), blk, 0, stream>>>(kbf, wkb, bk, kproj,  Kd);
  gemm_bf16_k<2><<<dim3(Ed / 64, (Bd * Kd) / 64), blk, 0, stream>>>(vbf, wvb, bv, vTp,    Kd);
  gemm_bf16_k<1><<<dim3(Ed / 64, (Bd * Md) / 64), blk, 0, stream>>>(mbf, wkb, bk, mkproj, Md);
  gemm_bf16_k<2><<<dim3(Ed / 64, (Bd * Md) / 64), blk, 0, stream>>>(mbf, wvb, bv, mvTp,   Md);

  // flash attention over K+M keys
  attn_flash_k<<<dim3(Qd / 16, Hd, Bd), dim3(32), 0, stream>>>(qproj, kproj, vTp, mkproj, mvTp, attnb);

  // output projection -> f32 d_out (B,Q,E)
  gemm_bf16_k<0><<<dim3(Ed / 64, (Bd * Qd) / 64), blk, 0, stream>>>(attnb, wob, bo, d_out, Qd);
}
